// SingleHeadAttention_12910671691923
// MI455X (gfx1250) — matrix operand order repaired and checked by reading, not compiled, not run on hardware
//
#include <hip/hip_runtime.h>
#include <hip/hip_bf16.h>

typedef __attribute__((ext_vector_type(16))) _Float16 v16h;
typedef __attribute__((ext_vector_type(8)))  float    v8f;
typedef unsigned int __attribute__((ext_vector_type(4))) uint32x4;
typedef int          __attribute__((ext_vector_type(4))) int32x4;
typedef int          __attribute__((ext_vector_type(8))) int32x8;

#define NB   4
#define SEQ  2048
#define HID  1024
#define DIM  1024

// Tile: 128 (M) x 256 (N) per workgroup, K-chunks of 32.
// 8 wave32s arranged 2x4; each wave owns a 64x64 sub-tile = 16 WMMA accs.
#define LDK 40   // 32 + 8 f16 pad -> 80B row pitch (16B aligned, TDM-pad compatible)

// ---------------------------------------------------------------------------
// WMMA: D = A(16x32 f16) * B(32x16 f16) + C(16x16 f32)
// ---------------------------------------------------------------------------
__device__ inline v8f wmma_f16(v16h a, v16h b, v8f c) {
    return __builtin_amdgcn_wmma_f32_16x16x32_f16(
        false, a, false, b, (short)0, c, false, false);
}

// A fragment from natural [row][k] LDS tile.  ISA layout: lane half h reads
// K{h*8..h*8+7} and K{16+h*8..16+h*8+7} -> two ds_load_b128.
__device__ inline v16h load_frag_a(const _Float16* base, int row16, int lane) {
    const _Float16* p = base + (size_t)(row16 + (lane & 15)) * LDK + ((lane >> 4) << 3);
    union { v16h v; uint4 q[2]; } u;
    u.q[0] = *(const uint4*)(p);
    u.q[1] = *(const uint4*)(p + 16);
    return u.v;
}
// B fragment from [n][k] LDS tile.  Lane half h reads K{h*8..h*8+7} and K{16+h*8..16+h*8+7}.
__device__ inline v16h load_frag_b(const _Float16* base, int col16, int lane) {
    const _Float16* p = base + (size_t)(col16 + (lane & 15)) * LDK + ((lane >> 4) << 3);
    union { v16h v; uint4 q[2]; } u;
    u.q[0] = *(const uint4*)(p);
    u.q[1] = *(const uint4*)(p + 16);
    return u.v;
}

union H4 { uint2 u2; _Float16 h[4]; };

// ---------------------------------------------------------------------------
// CDNA5 async datapath helpers
// ---------------------------------------------------------------------------
// Per-lane 16B global -> LDS async copy (ASYNCcnt).
__device__ inline void async_copy_b128(unsigned int lds_addr, const _Float16* g) {
    asm volatile("global_load_async_to_lds_b128 %0, %1, off"
                 :: "v"(lds_addr), "v"((unsigned long long)(uintptr_t)g)
                 : "memory");
}
__device__ inline void wait_async0() {
    asm volatile("s_wait_asynccnt 0x0" ::: "memory");
}

// TDM: one descriptor moves a (tile_h x tile_w) f16 tile from a row-major
// tensor (row pitch = tensor_w elems) into LDS with +16B pad every 64B row
// (pad_interval=3 -> 16 dwords, pad_amount=3 -> 4 dwords) == LDK pitch.
__device__ inline void tdm_load_tile_f16(unsigned int lds_addr, const _Float16* gptr,
                                         unsigned int tensor_w, unsigned int tensor_h,
                                         unsigned int tile_w, unsigned int tile_h) {
    const unsigned long long ga = (unsigned long long)(uintptr_t)gptr;
    uint32x4 g0;
    g0.x = 1u;                                   // count=1, user descriptor
    g0.y = lds_addr;                             // LDS destination (bytes)
    g0.z = (unsigned int)ga;                     // global_addr[31:0]
    g0.w = (unsigned int)(ga >> 32) | (2u << 30);// global_addr[56:32] | type=2
    int32x8 g1;
    g1[0] = (1 << 16) | (1 << 20) | (3 << 22) | (3 << 25);  // 2B elems, pad 4dw/16dw
    g1[1] = (int)(tensor_w << 16);                           // tensor_dim0[15:0]
    g1[2] = (int)((tensor_w >> 16) | (tensor_h << 16));      // dim0 hi | dim1 lo
    g1[3] = (int)((tensor_h >> 16) | (tile_w << 16));        // dim1 hi | tile_dim0
    g1[4] = (int)(tile_h & 0xFFFF);                          // tile_dim1
    g1[5] = (int)tensor_w;                                   // tensor_dim0_stride lo
    g1[6] = 0;
    g1[7] = 0;
    const int32x4 gz = {0, 0, 0, 0};
#if __clang_major__ >= 23
    const int32x8 gz8 = {0, 0, 0, 0, 0, 0, 0, 0};
    __builtin_amdgcn_tensor_load_to_lds(g0, g1, gz, gz, gz8, 0);
#else
    __builtin_amdgcn_tensor_load_to_lds(g0, g1, gz, gz, 0);
#endif
}

// ---------------------------------------------------------------------------
// Kernel 1: QKV projection  Y = X @ W  (M=8192, K=1024, N=1024), z -> q/k/v.
// Q,K stored [token][d] f16 (natural); V stored transposed [b][d][s] f16 so
// both attention GEMMs read their B tiles as straight row-major copies.
// ---------------------------------------------------------------------------
__global__ void __launch_bounds__(256)
qkv_gemm_kernel(const float* __restrict__ X,
                const float* __restrict__ Wq,
                const float* __restrict__ Wk,
                const float* __restrict__ Wv,
                _Float16* __restrict__ Qo,
                _Float16* __restrict__ Ko,
                _Float16* __restrict__ Vt)
{
    const int z  = blockIdx.z;
    const float* __restrict__ W = (z == 0) ? Wq : ((z == 1) ? Wk : Wv);
    const int n0 = blockIdx.x * 256;
    const int m0 = blockIdx.y * 128;

    __shared__ __align__(16) _Float16 As[128][LDK];   // [m][k]
    __shared__ __align__(16) _Float16 Bt[256][LDK];   // [n][k]

    const int tid   = threadIdx.x;
    const int lane  = tid & 31;
    const int wave  = tid >> 5;
    const int waveM = wave >> 2;
    const int waveN = wave & 3;

    v8f acc[4][4] = {};

    for (int k0 = 0; k0 < HID; k0 += 32) {
        if (k0 + 32 < HID) {
            __builtin_prefetch(X + (size_t)(m0 + (tid >> 1)) * HID + k0 + 32, 0, 3);
            __builtin_prefetch(W + (size_t)(k0 + 32 + (tid >> 3)) * DIM + n0, 0, 3);
        }
        // A tile: 128x32 fp32 -> f16 (natural layout, 8B packed stores)
#pragma unroll
        for (int i = 0; i < 4; ++i) {
            const int idx = tid + i * 256;
            const int row = idx >> 3;
            const int c4  = (idx & 7) * 4;
            const float4 f = *(const float4*)(X + (size_t)(m0 + row) * HID + k0 + c4);
            H4 t;
            t.h[0] = (_Float16)f.x; t.h[1] = (_Float16)f.y;
            t.h[2] = (_Float16)f.z; t.h[3] = (_Float16)f.w;
            *(uint2*)&As[row][c4] = t.u2;
        }
        // B tile: 32x256 fp32 -> f16, transposed scatter into [n][k]
#pragma unroll
        for (int i = 0; i < 8; ++i) {
            const int idx = tid + i * 256;
            const int kr  = idx >> 6;
            const int c4  = (idx & 63) * 4;
            const float4 f = *(const float4*)(W + (size_t)(k0 + kr) * DIM + n0 + c4);
            Bt[c4 + 0][kr] = (_Float16)f.x;
            Bt[c4 + 1][kr] = (_Float16)f.y;
            Bt[c4 + 2][kr] = (_Float16)f.z;
            Bt[c4 + 3][kr] = (_Float16)f.w;
        }
        __syncthreads();

        v16h af[4], bf[4];
#pragma unroll
        for (int mi = 0; mi < 4; ++mi) af[mi] = load_frag_a(&As[0][0], waveM * 64 + mi * 16, lane);
#pragma unroll
        for (int ni = 0; ni < 4; ++ni) bf[ni] = load_frag_b(&Bt[0][0], waveN * 64 + ni * 16, lane);
#pragma unroll
        for (int mi = 0; mi < 4; ++mi)
#pragma unroll
            for (int ni = 0; ni < 4; ++ni)
                acc[mi][ni] = wmma_f16(af[mi], bf[ni], acc[mi][ni]);
        __syncthreads();
    }

    const int half = lane >> 4;
    const int cl   = lane & 15;
#pragma unroll
    for (int mi = 0; mi < 4; ++mi)
#pragma unroll
        for (int ni = 0; ni < 4; ++ni)
#pragma unroll
            for (int r = 0; r < 8; ++r) {
                const int row = m0 + waveM * 64 + mi * 16 + r + 8 * half;
                const int col = n0 + waveN * 64 + ni * 16 + cl;
                const _Float16 h = (_Float16)acc[mi][ni][r];
                if (z == 0) {
                    Qo[(size_t)row * DIM + col] = h;
                } else if (z == 1) {
                    Ko[(size_t)row * DIM + col] = h;
                } else {
                    const int b = row >> 11, s = row & (SEQ - 1);
                    Vt[((size_t)b * DIM + col) * SEQ + s] = h;     // V^T: [b][d][s]
                }
            }
}

// ---------------------------------------------------------------------------
// Kernel 2: scores = (Q @ K^T) / sqrt(D), per batch.  Causal block-skip.
// A tile via TDM (tensor_load_to_lds), B tile via per-lane async-to-LDS.
// ---------------------------------------------------------------------------
__global__ void __launch_bounds__(256)
scores_gemm_kernel(const _Float16* __restrict__ Qf,
                   const _Float16* __restrict__ Kf,
                   float* __restrict__ Sc)
{
    if (2 * blockIdx.x > blockIdx.y) return;   // key tile fully above diagonal
    const int n0 = blockIdx.x * 256;           // keys
    const int m0 = blockIdx.y * 128;           // queries
    const int bb = blockIdx.z;

    __shared__ __align__(16) _Float16 As[128][LDK];   // [q][d]
    __shared__ __align__(16) _Float16 Bt[256][LDK];   // [key][d]

    const int tid   = threadIdx.x;
    const int lane  = tid & 31;
    const int wave  = tid >> 5;
    const int waveM = wave >> 2;
    const int waveN = wave & 3;

    v8f acc[4][4] = {};

    for (int k0 = 0; k0 < DIM; k0 += 32) {
        // A tile 128x32: single TDM descriptor, hardware pads rows to LDK pitch
        if (wave == 0) {
            tdm_load_tile_f16((unsigned int)(uintptr_t)&As[0][0],
                              Qf + ((size_t)bb * SEQ + m0) * DIM + k0,
                              DIM, NB * SEQ, 32, 128);
        }
        // B tile 256x32: K is stored naturally [key][d] -> straight 16B copies
#pragma unroll
        for (int i = 0; i < 4; ++i) {
            const int idx = tid + i * 256;
            const int n   = idx >> 2;
            const int c8  = (idx & 3) * 8;
            async_copy_b128((unsigned int)(uintptr_t)&Bt[n][c8],
                            Kf + ((size_t)bb * SEQ + n0 + n) * DIM + k0 + c8);
        }
        wait_async0();
        if (wave == 0) __builtin_amdgcn_s_wait_tensorcnt(0);
        __syncthreads();

        v16h af[4], bf[4];
#pragma unroll
        for (int mi = 0; mi < 4; ++mi) af[mi] = load_frag_a(&As[0][0], waveM * 64 + mi * 16, lane);
#pragma unroll
        for (int ni = 0; ni < 4; ++ni) bf[ni] = load_frag_b(&Bt[0][0], waveN * 64 + ni * 16, lane);
#pragma unroll
        for (int mi = 0; mi < 4; ++mi)
#pragma unroll
            for (int ni = 0; ni < 4; ++ni)
                acc[mi][ni] = wmma_f16(af[mi], bf[ni], acc[mi][ni]);
        __syncthreads();
    }

    const float scale = 0.03125f;   // 1/sqrt(1024)
    const int half = lane >> 4;
    const int cl   = lane & 15;
#pragma unroll
    for (int mi = 0; mi < 4; ++mi)
#pragma unroll
        for (int ni = 0; ni < 4; ++ni)
#pragma unroll
            for (int r = 0; r < 8; ++r) {
                const int row = m0 + waveM * 64 + mi * 16 + r + 8 * half;
                const int col = n0 + waveN * 64 + ni * 16 + cl;
                Sc[((size_t)bb * SEQ + row) * SEQ + col] = acc[mi][ni][r] * scale;
            }
}

// ---------------------------------------------------------------------------
// Kernel 3: causal softmax, in-place (zeros past the diagonal -> dense PV).
// ---------------------------------------------------------------------------
__global__ void __launch_bounds__(256)
softmax_kernel(float* __restrict__ Sc)
{
    const int q  = blockIdx.x;
    const int bb = blockIdx.y;
    float* __restrict__ row = Sc + ((size_t)bb * SEQ + q) * SEQ;

    const int tid  = threadIdx.x;
    const int lane = tid & 31;
    const int wave = tid >> 5;
    __shared__ float red[8];

    float m = -INFINITY;
    for (int k = tid; k <= q; k += 256) m = fmaxf(m, row[k]);
#pragma unroll
    for (int off = 16; off > 0; off >>= 1) m = fmaxf(m, __shfl_xor(m, off, 32));
    if (lane == 0) red[wave] = m;
    __syncthreads();
    m = red[0];
#pragma unroll
    for (int i = 1; i < 8; ++i) m = fmaxf(m, red[i]);
    __syncthreads();

    float s = 0.0f;
    for (int k = tid; k <= q; k += 256) s += __expf(row[k] - m);
#pragma unroll
    for (int off = 16; off > 0; off >>= 1) s += __shfl_xor(s, off, 32);
    if (lane == 0) red[wave] = s;
    __syncthreads();
    float tot = 0.0f;
#pragma unroll
    for (int i = 0; i < 8; ++i) tot += red[i];
    const float inv = 1.0f / tot;

    for (int k = tid; k < SEQ; k += 256)
        row[k] = (k <= q) ? __expf(row[k] - m) * inv : 0.0f;
}

// ---------------------------------------------------------------------------
// Kernel 4: out = P @ V per batch (K clipped at the diagonal block).
// B tile (from V^T [b][d][s]) via per-lane async-to-LDS.
// ---------------------------------------------------------------------------
__global__ void __launch_bounds__(256)
out_gemm_kernel(const float* __restrict__ P,
                const _Float16* __restrict__ Vt,
                float* __restrict__ Out)
{
    const int n0 = blockIdx.x * 256;
    const int m0 = blockIdx.y * 128;
    const int bb = blockIdx.z;

    __shared__ __align__(16) _Float16 As[128][LDK];   // [q][key]
    __shared__ __align__(16) _Float16 Bt[256][LDK];   // [d][key]

    const int tid   = threadIdx.x;
    const int lane  = tid & 31;
    const int wave  = tid >> 5;
    const int waveM = wave >> 2;
    const int waveN = wave & 3;

    v8f acc[4][4] = {};

    const int kend = (m0 + 128 < SEQ) ? (m0 + 128) : SEQ;   // P == 0 past diagonal
    for (int k0 = 0; k0 < kend; k0 += 32) {
        if (k0 + 32 < kend)
            __builtin_prefetch(P + ((size_t)bb * SEQ + m0 + (tid >> 1)) * SEQ + k0 + 32, 0, 3);
        // A tile: 128x32 fp32 P -> f16 (natural layout)
#pragma unroll
        for (int i = 0; i < 4; ++i) {
            const int idx = tid + i * 256;
            const int row = idx >> 3;
            const int c4  = (idx & 7) * 4;
            const float4 f = *(const float4*)(P + ((size_t)bb * SEQ + m0 + row) * SEQ + k0 + c4);
            H4 t;
            t.h[0] = (_Float16)f.x; t.h[1] = (_Float16)f.y;
            t.h[2] = (_Float16)f.z; t.h[3] = (_Float16)f.w;
            *(uint2*)&As[row][c4] = t.u2;
        }
        // B tile 256x32 from V^T: straight 16B async copies
#pragma unroll
        for (int i = 0; i < 4; ++i) {
            const int idx = tid + i * 256;
            const int d   = idx >> 2;
            const int c8  = (idx & 3) * 8;
            async_copy_b128((unsigned int)(uintptr_t)&Bt[d][c8],
                            Vt + ((size_t)bb * DIM + n0 + d) * SEQ + k0 + c8);
        }
        wait_async0();
        __syncthreads();

        v16h af[4], bf[4];
#pragma unroll
        for (int mi = 0; mi < 4; ++mi) af[mi] = load_frag_a(&As[0][0], waveM * 64 + mi * 16, lane);
#pragma unroll
        for (int ni = 0; ni < 4; ++ni) bf[ni] = load_frag_b(&Bt[0][0], waveN * 64 + ni * 16, lane);
#pragma unroll
        for (int mi = 0; mi < 4; ++mi)
#pragma unroll
            for (int ni = 0; ni < 4; ++ni)
                acc[mi][ni] = wmma_f16(af[mi], bf[ni], acc[mi][ni]);
        __syncthreads();
    }

    const int half = lane >> 4;
    const int cl   = lane & 15;
#pragma unroll
    for (int mi = 0; mi < 4; ++mi)
#pragma unroll
        for (int ni = 0; ni < 4; ++ni)
#pragma unroll
            for (int r = 0; r < 8; ++r) {
                const int row = m0 + waveM * 64 + mi * 16 + r + 8 * half;
                const int col = n0 + waveN * 64 + ni * 16 + cl;
                Out[((size_t)bb * SEQ + row) * DIM + col] = acc[mi][ni][r];
            }
}

// ---------------------------------------------------------------------------
// Launch
// ---------------------------------------------------------------------------
extern "C" void kernel_launch(void* const* d_in, const int* in_sizes, int n_in,
                              void* d_out, int out_size, void* d_ws, size_t ws_size,
                              hipStream_t stream) {
    const float* X  = (const float*)d_in[0];
    const float* Wq = (const float*)d_in[1];
    const float* Wk = (const float*)d_in[2];
    const float* Wv = (const float*)d_in[3];
    float* Out = (float*)d_out;

    // Workspace (112 MB): Q f16 | K f16 | V^T f16 | scores/P fp32
    char* ws = (char*)d_ws;
    _Float16* Qf = (_Float16*)(ws);
    _Float16* Kf = (_Float16*)(ws + ((size_t)16 << 20));
    _Float16* Vt = (_Float16*)(ws + ((size_t)32 << 20));
    float*    Sc = (float*)   (ws + ((size_t)48 << 20));

    qkv_gemm_kernel<<<dim3(DIM / 256, (NB * SEQ) / 128, 3), 256, 0, stream>>>(
        X, Wq, Wk, Wv, Qf, Kf, Vt);

    scores_gemm_kernel<<<dim3(SEQ / 256, SEQ / 128, NB), 256, 0, stream>>>(
        Qf, Kf, Sc);

    softmax_kernel<<<dim3(SEQ, NB), 256, 0, stream>>>(Sc);

    out_gemm_kernel<<<dim3(DIM / 256, SEQ / 128, NB), 256, 0, stream>>>(
        Sc, Vt, Out);
}
